// AttentionLayer_69904887710034
// MI455X (gfx1250) — matrix operand order repaired and checked by reading, not compiled, not run on hardware
//
#include <hip/hip_runtime.h>

typedef __bf16 bf16;
typedef __bf16 v16bf __attribute__((ext_vector_type(16)));
typedef __bf16 v8bf  __attribute__((ext_vector_type(8)));
typedef __bf16 v4bf  __attribute__((ext_vector_type(4)));
typedef float  v8f   __attribute__((ext_vector_type(8)));
typedef float  v4f   __attribute__((ext_vector_type(4)));

static constexpr int BN = 16;    // batch
static constexpr int SN = 1024;  // source
static constexpr int TN = 1024;  // target
static constexpr int HN = 1024;  // hidden
static constexpr int H2 = 2 * HN;

// ---------------------------------------------------------------------------
// CDNA5 async global->LDS copy (ASYNCcnt-tracked, bypasses VGPR staging).
// lds_off = low 32 bits of a generic pointer to LDS == LDS byte address
// (see ISA section 10.2).
// ---------------------------------------------------------------------------
__device__ __forceinline__ unsigned lds_addr(const void* p) {
  return (unsigned)(unsigned long long)(uintptr_t)p;
}
__device__ __forceinline__ void async_b128(unsigned lds_off, const bf16* g) {
  asm volatile("global_load_async_to_lds_b128 %0, %1, off"
               :: "v"(lds_off), "v"(g) : "memory");
}
__device__ __forceinline__ void wait_async4() {
  asm volatile("s_wait_asynccnt 4" ::: "memory");
}
__device__ __forceinline__ void wait_async0() {
  asm volatile("s_wait_asynccnt 0" ::: "memory");
}

// ---------------------------------------------------------------------------
// Kernel A: fp32 -> bf16 cast (8 elements / thread)
// ---------------------------------------------------------------------------
__global__ __launch_bounds__(256) void cast_bf16_kernel(
    const float* __restrict__ in, bf16* __restrict__ out, int n8) {
  int i = blockIdx.x * blockDim.x + threadIdx.x;
  if (i >= n8) return;
  size_t base = (size_t)i * 8;
  v4f a = *(const v4f*)(in + base);
  v4f b = *(const v4f*)(in + base + 4);
  v8bf o;
#pragma unroll
  for (int j = 0; j < 4; ++j) { o[j] = (bf16)a[j]; o[j + 4] = (bf16)b[j]; }
  *(v8bf*)(out + base) = o;
}

// ---------------------------------------------------------------------------
// Kernel T: enc fp32 [b][s][h] -> bf16 encT [b][h][s] (32x32 LDS tiles)
// ---------------------------------------------------------------------------
__global__ __launch_bounds__(256) void transpose_cast_kernel(
    const float* __restrict__ in, bf16* __restrict__ outT) {
  const int b  = blockIdx.z;
  const int s0 = blockIdx.y * 32;
  const int h0 = blockIdx.x * 32;
  __shared__ float tile[32][33];
  const int tr = threadIdx.x >> 3;        // 0..31
  const int tc = (threadIdx.x & 7) * 4;   // 0,4,..,28
  v4f v = *(const v4f*)(in + ((size_t)b * SN + s0 + tr) * HN + h0 + tc);
#pragma unroll
  for (int j = 0; j < 4; ++j) tile[tr][tc + j] = v[j];
  __syncthreads();
  v4bf o;
#pragma unroll
  for (int j = 0; j < 4; ++j) o[j] = (bf16)tile[tc + j][tr];
  *(v4bf*)(outT + ((size_t)b * HN + h0 + tr) * SN + s0 + tc) = o;
}

// ---------------------------------------------------------------------------
// Kernel B: score = enc @ dec^T for one (b, 64-wide t tile); flash softmax
// stats over s; raw scores stored t-major: score[(b*T + t)*S + s].
// Double-buffered LDS, async global->LDS fills overlapped with WMMA.
// ---------------------------------------------------------------------------
__global__ __launch_bounds__(128) void score_stats_kernel(
    const bf16* __restrict__ enc, const bf16* __restrict__ dec,
    float* __restrict__ score, float* __restrict__ smax,
    float* __restrict__ sinv) {
  const int b    = blockIdx.y;
  const int t0   = blockIdx.x * 64;
  const int tid  = threadIdx.x;
  const int lane = tid & 31;
  const int wave = tid >> 5;
  const int ln   = lane & 15;
  const int hi   = lane >> 4;

  __shared__ __align__(32) bf16 sA[2][64 * 32];  // enc tile [s][k]
  __shared__ __align__(32) bf16 sB[2][64 * 32];  // dec tile [t][k]

  const int r    = tid >> 1;   // staging row 0..63
  const int half = tid & 1;    // 16-element half
  const bf16* encRow = enc + ((size_t)b * SN + r) * HN + half * 16;
  const bf16* decRow = dec + ((size_t)b * TN + t0 + r) * HN + half * 16;
  unsigned ldsA[2], ldsB[2];
#pragma unroll
  for (int u = 0; u < 2; ++u) {
    ldsA[u] = lds_addr(&sA[u][r * 32 + half * 16]);
    ldsB[u] = lds_addr(&sB[u][r * 32 + half * 16]);
  }

  float m_run = -3.0e38f;
  float l_run = 0.0f;
  constexpr int KSTEPS = HN / 32;

  for (int s0 = 0; s0 < SN; s0 += 64) {
    v8f acc[4];
#pragma unroll
    for (int st = 0; st < 4; ++st)
#pragma unroll
      for (int j = 0; j < 8; ++j) acc[st][j] = 0.0f;

    const bf16* encS = encRow + (size_t)s0 * HN;
    // prologue: fill buffer 0 (k0 = 0)
#pragma unroll
    for (int c = 0; c < 2; ++c) {
      async_b128(ldsA[0] + c * 16, encS + c * 8);
      async_b128(ldsB[0] + c * 16, decRow + c * 8);
    }

    for (int ks = 0; ks < KSTEPS; ++ks) {
      const int cur = ks & 1;
      if (ks + 1 < KSTEPS) {
        const int k1 = (ks + 1) * 32;
#pragma unroll
        for (int c = 0; c < 2; ++c) {
          async_b128(ldsA[cur ^ 1] + c * 16, encS + k1 + c * 8);
          async_b128(ldsB[cur ^ 1] + c * 16, decRow + k1 + c * 8);
        }
        wait_async4();   // current buffer's 4 fills complete (in-order)
      } else {
        wait_async0();
      }
      __syncthreads();

      v16bf fb = *(const v16bf*)(&sB[cur][(wave * 16 + ln) * 32 + hi * 16]);
#pragma unroll
      for (int st = 0; st < 4; ++st) {
        const bf16* ap = &sA[cur][(st * 16 + ln) * 32 + hi * 16];
        v8bf lo  = *(const v8bf*)(ap);
        v8bf hi8 = *(const v8bf*)(ap + 8);
        v16bf fa;
#pragma unroll
        for (int j = 0; j < 8; ++j) { fa[j] = lo[j]; fa[j + 8] = hi8[j]; }
        acc[st] = __builtin_amdgcn_wmma_f32_16x16x32_bf16(
            false, fa, false, fb, (short)0, acc[st], false, false);
      }
      __syncthreads();  // readers done before this buffer is refilled
    }

    // --- online softmax stats for this 64-row s tile (per column n=ln) ---
    float tmax = -3.0e38f;
#pragma unroll
    for (int st = 0; st < 4; ++st)
#pragma unroll
      for (int j = 0; j < 8; ++j) tmax = fmaxf(tmax, acc[st][j]);
    tmax = fmaxf(tmax, __shfl_xor(tmax, 16, 32));
    float m_new = fmaxf(m_run, tmax);
    float psum = 0.0f;
#pragma unroll
    for (int st = 0; st < 4; ++st)
#pragma unroll
      for (int j = 0; j < 8; ++j) psum += __expf(acc[st][j] - m_new);
    psum += __shfl_xor(psum, 16, 32);
    l_run = l_run * __expf(m_run - m_new) + psum;
    m_run = m_new;

    // --- store raw scores t-major; lane's 8 values are s-contiguous ---
    float* sp = score + ((size_t)b * TN + t0 + wave * 16 + ln) * SN + s0;
#pragma unroll
    for (int st = 0; st < 4; ++st) {
      v4f w0, w1;
#pragma unroll
      for (int j = 0; j < 4; ++j) { w0[j] = acc[st][j]; w1[j] = acc[st][4 + j]; }
      *(v4f*)(sp + st * 16 + hi * 8)     = w0;
      *(v4f*)(sp + st * 16 + hi * 8 + 4) = w1;
    }
  }

  if (hi == 0) {
    size_t t = (size_t)b * TN + t0 + wave * 16 + ln;
    smax[t] = m_run;
    sinv[t] = 1.0f / l_run;
  }
}

// ---------------------------------------------------------------------------
// Kernel P: p[(b*T+t)*S+s] = bf16(exp(score - max[t]) * inv_sum[t])
// ---------------------------------------------------------------------------
__global__ __launch_bounds__(256) void pgen_kernel(
    const float* __restrict__ score, const float* __restrict__ smax,
    const float* __restrict__ sinv, bf16* __restrict__ p, int n8) {
  int i = blockIdx.x * blockDim.x + threadIdx.x;
  if (i >= n8) return;
  size_t e  = (size_t)i * 8;
  size_t bt = e >> 10;  // / SN
  const float mx = smax[bt];
  const float iv = sinv[bt];
  v4f a = *(const v4f*)(score + e);
  v4f b = *(const v4f*)(score + e + 4);
  v8bf o;
#pragma unroll
  for (int j = 0; j < 4; ++j) {
    o[j]     = (bf16)(__expf(a[j] - mx) * iv);
    o[j + 4] = (bf16)(__expf(b[j] - mx) * iv);
  }
  *(v8bf*)(p + e) = o;
}

// ---------------------------------------------------------------------------
// Kernel C: encoder_vector[b,t,h] = sum_s p[t,s]*encT[h,s].
// A = P (M=t, K=s) row-contiguous; B = encT (K=s, N=h) row-contiguous.
// Both tiles async-filled into double-buffered LDS; inner loop is pure
// async-issue -> wait -> barrier -> fragment loads -> WMMA.
// ---------------------------------------------------------------------------
__global__ __launch_bounds__(128) void av_kernel(
    const bf16* __restrict__ p, const bf16* __restrict__ encT,
    float* __restrict__ out) {
  const int b    = blockIdx.z;
  const int t0   = blockIdx.y * 64;
  const int h0   = blockIdx.x * 64;
  const int tid  = threadIdx.x;
  const int lane = tid & 31;
  const int wave = tid >> 5;
  const int ln   = lane & 15;
  const int hi   = lane >> 4;

  __shared__ __align__(32) bf16 sP[2][64 * 32];  // [t][s]
  __shared__ __align__(32) bf16 sV[2][64 * 32];  // [h][s]

  const int r    = tid >> 1;
  const int half = tid & 1;
  const bf16* pRow = p    + ((size_t)b * TN + t0 + r) * SN + half * 16;
  const bf16* vRow = encT + ((size_t)b * HN + h0 + r) * SN + half * 16;
  unsigned ldsP[2], ldsV[2];
#pragma unroll
  for (int u = 0; u < 2; ++u) {
    ldsP[u] = lds_addr(&sP[u][r * 32 + half * 16]);
    ldsV[u] = lds_addr(&sV[u][r * 32 + half * 16]);
  }

  v8f acc[4];
#pragma unroll
  for (int st = 0; st < 4; ++st)
#pragma unroll
    for (int j = 0; j < 8; ++j) acc[st][j] = 0.0f;

  constexpr int SSTEPS = SN / 32;
  // prologue: fill buffer 0 (s0 = 0)
#pragma unroll
  for (int c = 0; c < 2; ++c) {
    async_b128(ldsP[0] + c * 16, pRow + c * 8);
    async_b128(ldsV[0] + c * 16, vRow + c * 8);
  }

  for (int ss = 0; ss < SSTEPS; ++ss) {
    const int cur = ss & 1;
    if (ss + 1 < SSTEPS) {
      const int s1 = (ss + 1) * 32;
#pragma unroll
      for (int c = 0; c < 2; ++c) {
        async_b128(ldsP[cur ^ 1] + c * 16, pRow + s1 + c * 8);
        async_b128(ldsV[cur ^ 1] + c * 16, vRow + s1 + c * 8);
      }
      wait_async4();
    } else {
      wait_async0();
    }
    __syncthreads();

    v16bf fb = *(const v16bf*)(&sV[cur][(wave * 16 + ln) * 32 + hi * 16]);
#pragma unroll
    for (int st = 0; st < 4; ++st) {
      const bf16* ap = &sP[cur][(st * 16 + ln) * 32 + hi * 16];
      v8bf lo  = *(const v8bf*)(ap);
      v8bf hi8 = *(const v8bf*)(ap + 8);
      v16bf fa;
#pragma unroll
      for (int j = 0; j < 8; ++j) { fa[j] = lo[j]; fa[j + 8] = hi8[j]; }
      acc[st] = __builtin_amdgcn_wmma_f32_16x16x32_bf16(
          false, fa, false, fb, (short)0, acc[st], false, false);
    }
    __syncthreads();
  }

  // D: M = t (j + 8*hi within subtile), N = h (ln). Concat second half.
#pragma unroll
  for (int st = 0; st < 4; ++st)
#pragma unroll
    for (int j = 0; j < 8; ++j) {
      int t = t0 + st * 16 + j + 8 * hi;
      out[((size_t)b * TN + t) * H2 + HN + h0 + wave * 16 + ln] = acc[st][j];
    }
}

// ---------------------------------------------------------------------------
// Kernel D: out[b,t,0:H] = dec[b,t,:]
// ---------------------------------------------------------------------------
__global__ __launch_bounds__(256) void copy_dec_kernel(
    const float* __restrict__ dec, float* __restrict__ out, int n4) {
  int i = blockIdx.x * blockDim.x + threadIdx.x;
  if (i >= n4) return;
  size_t e  = (size_t)i * 4;
  size_t bt = e >> 10;          // / HN
  size_t h  = e & (HN - 1);
  *(v4f*)(out + bt * H2 + h) = *(const v4f*)(dec + e);
}

// ---------------------------------------------------------------------------
extern "C" void kernel_launch(void* const* d_in, const int* in_sizes, int n_in,
                              void* d_out, int out_size, void* d_ws,
                              size_t ws_size, hipStream_t stream) {
  (void)in_sizes; (void)n_in; (void)out_size; (void)ws_size;
  const float* enc = (const float*)d_in[0];  // [B,S,H]
  const float* dec = (const float*)d_in[1];  // [B,T,H]
  float* out = (float*)d_out;                // [B,T,2H]

  char* ws = (char*)d_ws;
  const size_t encBytes = (size_t)BN * SN * HN * sizeof(bf16);   // 32 MiB
  const size_t decBytes = (size_t)BN * TN * HN * sizeof(bf16);   // 32 MiB
  const size_t trBytes  = (size_t)BN * HN * SN * sizeof(bf16);   // 32 MiB
  const size_t scBytes  = (size_t)BN * TN * SN * sizeof(float);  // 64 MiB
  const size_t stBytes  = (size_t)BN * TN * sizeof(float);       // 64 KiB
  bf16*  encb  = (bf16*)ws;                                  // dead after B
  bf16*  pbuf  = (bf16*)ws;                                  // aliases encb
  bf16*  decb  = (bf16*)(ws + encBytes);
  bf16*  encT  = (bf16*)(ws + encBytes + decBytes);
  float* score = (float*)(ws + encBytes + decBytes + trBytes);
  float* stmax = (float*)(ws + encBytes + decBytes + trBytes + scBytes);
  float* stinv = (float*)(ws + encBytes + decBytes + trBytes + scBytes + stBytes);

  const int n8 = BN * SN * HN / 8;
  cast_bf16_kernel<<<n8 / 256, 256, 0, stream>>>(enc, encb, n8);
  cast_bf16_kernel<<<n8 / 256, 256, 0, stream>>>(dec, decb, n8);

  transpose_cast_kernel<<<dim3(HN / 32, SN / 32, BN), 256, 0, stream>>>(enc, encT);

  score_stats_kernel<<<dim3(TN / 64, BN), 128, 0, stream>>>(
      encb, decb, score, stmax, stinv);

  const int np8 = BN * TN * SN / 8;
  pgen_kernel<<<np8 / 256, 256, 0, stream>>>(score, stmax, stinv, pbuf, np8);

  av_kernel<<<dim3(HN / 64, TN / 64, BN), 128, 0, stream>>>(pbuf, encT, out);

  const int n4 = BN * TN * HN / 4;
  copy_dec_kernel<<<n4 / 256, 256, 0, stream>>>(dec, out, n4);
}
